// DSEModel_11768210391131
// MI455X (gfx1250) — compile-verified, matrix operand order checked by reading, not run on hardware
//
// DSEModel forward for MI455X (gfx1250), wave32 + WMMA.
//
// Design notes (compile-only environment; reasoning from MI455X specs):
//  * All matmul-shaped work (linears, GRU input projections, 1x1/3x3 convs via
//    im2col, GCN feature transform, attention projections, final drug@se^T)
//    runs through one tiled GEMM kernel using v_wmma_f32_16x16x32_bf16:
//    fp32 activations/weights are converted to bf16 while staged into LDS,
//    accumulation stays fp32. ~150 GFLOP total over a working set that fits
//    in the 192MB L2; HBM (23.3 TB/s) is not the limiter, the matrix units are.
//  * GEMM tile: 256 threads = 8 wave32 per block, 128x64 block tile, BK=32,
//    each wave computes a 32x32 tile (2x2 WMMA accumulators). LDS rows padded
//    to 40 halves so the 8-half fragment loads are 16B aligned (ds_read_b128).
//    Fragment packing follows the CDNA5 ISA 16-bit A/B VGPR layout tables.
//  * Staging uses an unconditional float4 fast path for interior tiles
//    (global_load_b128 + packed bf16 converts + ds_store_b64) and a branchless
//    clamped-select path for edge tiles, so loads batch instead of serializing
//    behind per-element exec-mask branches and waits.
//  * Elementwise/reduction tail (BN stats per column, GRU gates, S=3 softmax
//    attention, GCN scatter with atomics) on VALU kernels.
//  * d_out's 32M-float `outputs` region is used as scratch for large
//    transients (im2col buffers, ds, SE-branch buffers) -- all dead before the
//    final GEMM fills it. drug/se_emb are computed in place in d_out.
//  * ASSUMPTION: the params pytree is flattened leaf-wise in insertion order,
//    giving 141 device pointers (10 inputs, 129 param leaves, 2 edge indices).
//  * Workspace use: ~192 MB from d_ws (bump allocated below).

#include <hip/hip_runtime.h>
#include <hip/hip_bf16.h>
#include <math.h>

#define D_N 4000
#define S_N 8000

typedef __bf16 bf16_t;
typedef bf16_t bf16x4  __attribute__((ext_vector_type(4)));
typedef bf16_t bf16x8  __attribute__((ext_vector_type(8)));
typedef bf16_t bf16x16 __attribute__((ext_vector_type(16)));
typedef float  f32x4   __attribute__((ext_vector_type(4)));
typedef float  f32x8   __attribute__((ext_vector_type(8)));

__device__ __forceinline__ float dse_gelu(float x) {
  return 0.5f * x * (1.0f + erff(x * 0.70710678118654752f));
}
__device__ __forceinline__ float dse_sigmoid(float x) {
  return 1.0f / (1.0f + __expf(-x));
}

// ---------------------------------------------------------------------------
// WMMA GEMM: C[M,N] = act(A[M,K] @ B[N,K]^T + bias[N])   (act 0=none 1=gelu)
// ---------------------------------------------------------------------------
#define GBM 128
#define GBN 64
#define GBK 32
#define GPAD 8  // pad to 40 halves/row = 80B -> 16B-aligned 8-half loads

__global__ __launch_bounds__(256)
void dse_wmma_gemm(const float* __restrict__ A, int lda,
                   const float* __restrict__ B, int ldb,
                   float* __restrict__ C, int ldc,
                   int M, int N, int K,
                   const float* __restrict__ bias, int act)
{
  __shared__ __align__(16) bf16_t As[GBM][GBK + GPAD];
  __shared__ __align__(16) bf16_t Bs[GBN][GBK + GPAD];

  const int tid  = threadIdx.x;
  const int wid  = tid >> 5;
  const int lane = tid & 31;
  const int wm   = wid >> 1;   // 0..3 -> 32-row slab
  const int wn   = wid & 1;    // 0..1 -> 32-col slab
  const int l15  = lane & 15;
  const int hi   = lane >> 4;  // 0/1 : selects K half per ISA layout

  const int m0 = blockIdx.y * GBM;
  const int n0 = blockIdx.x * GBN;
  const bool mfull = (m0 + GBM <= M);
  const bool nfull = (n0 + GBN <= N);

  f32x8 acc[2][2];
#pragma unroll
  for (int r = 0; r < 2; ++r)
#pragma unroll
    for (int c = 0; c < 2; ++c)
#pragma unroll
      for (int e = 0; e < 8; ++e) acc[r][c][e] = 0.0f;

  const int nk = (K + GBK - 1) / GBK;
  for (int kt = 0; kt < nk; ++kt) {
    const int k0 = kt * GBK;
    const bool kfull = (k0 + GBK <= K);

    // ---- stage A (128x32) f32 -> bf16 ----
    if (mfull & kfull) {
      // interior tile: unconditional float4 loads (global_load_b128),
      // packed cvt, 8B LDS stores. 4 passes x 256 thr x 4 elems = 4096.
#pragma unroll
      for (int p = 0; p < 4; ++p) {
        int i = p * 256 + tid;
        int r = i >> 3, cg = (i & 7) << 2;
        f32x4 v = *(const f32x4*)(A + (size_t)(m0 + r) * lda + (k0 + cg));
        bf16x4 h;
#pragma unroll
        for (int j = 0; j < 4; ++j) h[j] = (bf16_t)v[j];
        *(bf16x4*)&As[r][cg] = h;
      }
    } else {
      // edge tile: branchless clamped loads + select-zero (no exec-mask
      // divergence around the loads, so they can all issue before waiting)
#pragma unroll
      for (int i = tid; i < GBM * GBK; i += 256) {
        int r = i >> 5, c = i & 31;
        int gm = m0 + r, gk = k0 + c;
        bool ok = (gm < M) && (gk < K);
        int cm = ok ? gm : (M - 1);
        int ck = ok ? gk : 0;
        float v = A[(size_t)cm * lda + ck];
        As[r][c] = (bf16_t)(ok ? v : 0.0f);
      }
    }
    // ---- stage B (64x32) f32 -> bf16 ----
    if (nfull & kfull) {
#pragma unroll
      for (int p = 0; p < 2; ++p) {
        int i = p * 256 + tid;
        int r = i >> 3, cg = (i & 7) << 2;
        f32x4 v = *(const f32x4*)(B + (size_t)(n0 + r) * ldb + (k0 + cg));
        bf16x4 h;
#pragma unroll
        for (int j = 0; j < 4; ++j) h[j] = (bf16_t)v[j];
        *(bf16x4*)&Bs[r][cg] = h;
      }
    } else {
#pragma unroll
      for (int i = tid; i < GBN * GBK; i += 256) {
        int r = i >> 5, c = i & 31;
        int gn = n0 + r, gk = k0 + c;
        bool ok = (gn < N) && (gk < K);
        int cn = ok ? gn : (N - 1);
        int ck = ok ? gk : 0;
        float v = B[(size_t)cn * ldb + ck];
        Bs[r][c] = (bf16_t)(ok ? v : 0.0f);
      }
    }
    // prefetch next A K-tile (gfx1250 global_prefetch_b8 path)
    if (kt + 1 < nk) {
      int pm = m0 + (tid >> 1);
      int pk = k0 + GBK + ((tid & 1) << 4);
      if (pm < M && pk < K) __builtin_prefetch(&A[(size_t)pm * lda + pk], 0, 1);
    }
    __syncthreads();

    // fragments per CDNA5 16-bit A/B layout: lane l -> row (l&15),
    // elem e -> K = hi*8 + e (e<8)  |  16 + hi*8 + (e-8)
    union Frag { bf16x16 v; bf16x8 h[2]; };
    Frag af[2], bfr[2];
#pragma unroll
    for (int r = 0; r < 2; ++r) {
      int am = wm * 32 + r * 16 + l15;
      af[r].h[0] = *(const bf16x8*)&As[am][hi * 8];
      af[r].h[1] = *(const bf16x8*)&As[am][16 + hi * 8];
    }
#pragma unroll
    for (int c = 0; c < 2; ++c) {
      int bn = wn * 32 + c * 16 + l15;
      bfr[c].h[0] = *(const bf16x8*)&Bs[bn][hi * 8];
      bfr[c].h[1] = *(const bf16x8*)&Bs[bn][16 + hi * 8];
    }
#pragma unroll
    for (int r = 0; r < 2; ++r)
#pragma unroll
      for (int c = 0; c < 2; ++c)
        acc[r][c] = __builtin_amdgcn_wmma_f32_16x16x32_bf16(
            false, af[r].v, false, bfr[c].v, (short)0, acc[r][c], false, false);
    __syncthreads();
  }

  // store: C/D layout -> elem e : M = e + 8*hi ; N = l15
#pragma unroll
  for (int r = 0; r < 2; ++r) {
#pragma unroll
    for (int c = 0; c < 2; ++c) {
      int n = n0 + wn * 32 + c * 16 + l15;
      if (n >= N) continue;
      float bv = bias ? bias[n] : 0.0f;
#pragma unroll
      for (int e = 0; e < 8; ++e) {
        int m = m0 + wm * 32 + r * 16 + hi * 8 + e;
        if (m >= M) continue;
        float v = acc[r][c][e] + bv;
        if (act == 1) v = dse_gelu(v);
        C[(size_t)m * ldc + n] = v;
      }
    }
  }
}

// ---------------------------------------------------------------------------
// BatchNorm over rows (per-column stats, biased var) + optional GELU.
// One block per column. Working set sits in L2, so strided reads are cheap.
// ---------------------------------------------------------------------------
__global__ __launch_bounds__(256)
void dse_colbn(const float* __restrict__ x, int ldx,
               float* __restrict__ y, int ldy,
               const float* __restrict__ g, const float* __restrict__ b,
               int M, int act)
{
  const int c = blockIdx.x;
  __shared__ float ss[256], sq[256];
  float s = 0.f, q = 0.f;
  for (int r = threadIdx.x; r < M; r += 256) {
    float v = x[(size_t)r * ldx + c];
    s += v; q += v * v;
  }
  ss[threadIdx.x] = s; sq[threadIdx.x] = q;
  __syncthreads();
  for (int o = 128; o > 0; o >>= 1) {
    if (threadIdx.x < o) {
      ss[threadIdx.x] += ss[threadIdx.x + o];
      sq[threadIdx.x] += sq[threadIdx.x + o];
    }
    __syncthreads();
  }
  const float mean = ss[0] / (float)M;
  const float var  = sq[0] / (float)M - mean * mean;
  const float rstd = rsqrtf(var + 1e-5f);
  const float gg = g[c], bb = b[c];
  for (int r = threadIdx.x; r < M; r += 256) {
    float v = (x[(size_t)r * ldx + c] - mean) * rstd * gg + bb;
    if (act) v = dse_gelu(v);
    y[(size_t)r * ldy + c] = v;
  }
}

// LayerNorm over the 64 features of each row (norm1).
__global__ void dse_rownorm64(const float* __restrict__ x, float* __restrict__ y,
                              const float* __restrict__ g, const float* __restrict__ b)
{
  const int row = blockIdx.x, c = threadIdx.x;  // 64 threads
  __shared__ float ss[64], sq[64];
  float v = x[(size_t)row * 64 + c];
  ss[c] = v; sq[c] = v * v;
  __syncthreads();
  for (int o = 32; o > 0; o >>= 1) {
    if (c < o) { ss[c] += ss[c + o]; sq[c] += sq[c + o]; }
    __syncthreads();
  }
  float mean = ss[0] * (1.f / 64.f);
  float var  = sq[0] * (1.f / 64.f) - mean * mean;
  y[(size_t)row * 64 + c] = g[c] * (v - mean) * rsqrtf(var + 1e-5f) + b[c];
}

// GRU gates from gpre = x@Wih^T + bih (h0 = 0): out = (1-z)*n
__global__ void dse_gru_act(const float* __restrict__ gpre, const float* __restrict__ bhh,
                            float* __restrict__ out, int M, int h, int ldo, int coff)
{
  int i = blockIdx.x * blockDim.x + threadIdx.x;
  if (i >= M * h) return;
  int m = i / h, j = i % h;
  const float* gr = gpre + (size_t)m * 3 * h;
  float r = dse_sigmoid(gr[j] + bhh[j]);
  float z = dse_sigmoid(gr[h + j] + bhh[h + j]);
  float n = tanhf(gr[2 * h + j] + r * bhh[2 * h + j]);
  out[(size_t)m * ldo + coff + j] = (1.0f - z) * n;
}

// ------------------------- elementwise helpers -----------------------------
__global__ void dse_mean4(const float* a, const float* b, const float* c,
                          const float* d, float* o, int n) {
  int i = blockIdx.x * blockDim.x + threadIdx.x;
  if (i < n) o[i] = (a[i] + b[i] + c[i] + d[i]) * 0.25f;
}
__global__ void dse_mean3(const float* a, const float* b, const float* c,
                          float* o, int n) {
  int i = blockIdx.x * blockDim.x + threadIdx.x;
  if (i < n) o[i] = (a[i] + b[i] + c[i]) * (1.f / 3.f);
}
__global__ void dse_avg2(const float* a, const float* b, float* o, int n) {
  int i = blockIdx.x * blockDim.x + threadIdx.x;
  if (i < n) o[i] = (a[i] + b[i]) * 0.5f;
}
__global__ void dse_add2(const float* a, const float* b, float* o, int n) {
  int i = blockIdx.x * blockDim.x + threadIdx.x;
  if (i < n) o[i] = a[i] + b[i];
}
__global__ void dse_zero(float* p, int n) {
  int i = blockIdx.x * blockDim.x + threadIdx.x;
  if (i < n) p[i] = 0.0f;
}

// mcn pack: X1[(n*4+l)*64+c] = {v,cses,f,mm}[l][n,c]
__global__ void dse_pack_mcn(const float* v, const float* cs, const float* f,
                             const float* mm, float* X1) {
  int i = blockIdx.x * blockDim.x + threadIdx.x;
  if (i >= D_N * 4 * 64) return;
  int c = i & 63, rl = i >> 6, n = rl >> 2, l = rl & 3;
  const float* src = (l == 0) ? v : (l == 1) ? cs : (l == 2) ? f : mm;
  X1[i] = src[(size_t)n * 64 + c];
}
// matk pack: rows s*D_N+b, s in {v,cses,f}
__global__ void dse_pack3(const float* v, const float* cs, const float* f, float* o) {
  int i = blockIdx.x * blockDim.x + threadIdx.x;
  if (i >= 3 * D_N * 64) return;
  int c = i & 63, r = i >> 6, s = r / D_N, b = r % D_N;
  const float* src = (s == 0) ? v : (s == 1) ? cs : f;
  o[i] = src[(size_t)b * 64 + c];
}

// im2col k=3 pad=1 within groups of length L; out col = c*3+dk matches [O,C,3] weights
__global__ void dse_im2col3(const float* __restrict__ in, float* __restrict__ out,
                            int rows, int L, int C) {
  int i = blockIdx.x * blockDim.x + threadIdx.x;
  if (i >= rows * C) return;
  int r = i / C, c = i % C, pos = r % L;
  float* o = out + (size_t)r * 3 * C + (size_t)c * 3;
  o[0] = (pos > 0)     ? in[(size_t)(r - 1) * C + c] : 0.f;
  o[1] =                 in[(size_t)r * C + c];
  o[2] = (pos < L - 1) ? in[(size_t)(r + 1) * C + c] : 0.f;
}
// avg_pool1d k=3 s=1 p=1 (always /3; zero boundary)
__global__ void dse_avgpool3(const float* __restrict__ in, float* __restrict__ out,
                             int rows, int L, int C) {
  int i = blockIdx.x * blockDim.x + threadIdx.x;
  if (i >= rows * C) return;
  int r = i / C, c = i % C, pos = r % L;
  float a = (pos > 0)     ? in[(size_t)(r - 1) * C + c] : 0.f;
  float b =                 in[(size_t)r * C + c];
  float d = (pos < L - 1) ? in[(size_t)(r + 1) * C + c] : 0.f;
  out[i] = (a + b + d) * (1.f / 3.f);
}
// [16000,256] branch -> ds[4000,4096] at channel offset coff: ds[n,(coff+o)*4+l]
__global__ void dse_repack_ds(const float* __restrict__ br, float* __restrict__ ds, int coff) {
  int i = blockIdx.x * blockDim.x + threadIdx.x;
  if (i >= D_N * 4 * 256) return;
  int o = i & 255, rl = i >> 8, n = rl >> 2, l = rl & 3;
  ds[(size_t)n * 4096 + (size_t)(coff + o) * 4 + l] = br[i];
}

// MHA with S=3 keys, 2 heads of 32: softmax over 3 + weighted sum
__global__ void dse_attn1(const float* __restrict__ qh, const float* __restrict__ kh,
                          const float* __restrict__ vh, float* __restrict__ ctx) {
  int t = blockIdx.x * blockDim.x + threadIdx.x;
  if (t >= D_N * 2) return;
  int b = t >> 1, h = t & 1, off = h * 32;
  float sc[3];
  for (int s = 0; s < 3; ++s) {
    float d = 0.f;
    for (int k = 0; k < 32; ++k)
      d += qh[(size_t)b * 64 + off + k] * kh[((size_t)s * D_N + b) * 64 + off + k];
    sc[s] = d * 0.17677669529663687f;  // 1/sqrt(32)
  }
  float mx = fmaxf(sc[0], fmaxf(sc[1], sc[2]));
  float e0 = __expf(sc[0] - mx), e1 = __expf(sc[1] - mx), e2 = __expf(sc[2] - mx);
  float inv = 1.f / (e0 + e1 + e2);
  for (int k = 0; k < 32; ++k) {
    float v = e0 * vh[((size_t)0 * D_N + b) * 64 + off + k] +
              e1 * vh[((size_t)1 * D_N + b) * 64 + off + k] +
              e2 * vh[((size_t)2 * D_N + b) * 64 + off + k];
    ctx[(size_t)b * 64 + off + k] = v * inv;
  }
}

// ------------------------------- GCN ---------------------------------------
__global__ void dse_gcn_deg(float* deg, const int* __restrict__ dst,
                            const float* __restrict__ w, int E) {
  int e = blockIdx.x * blockDim.x + threadIdx.x;
  if (e < E) atomicAdd(&deg[dst[e]], w[e]);
}
__global__ void dse_gcn_degself(float* deg, int n) {
  int i = blockIdx.x * blockDim.x + threadIdx.x;
  if (i < n) deg[i] += 1.0f;
}
__global__ void dse_gcn_dinv(float* dinv, const float* deg, int n) {
  int i = blockIdx.x * blockDim.x + threadIdx.x;
  if (i < n) { float d = deg[i]; dinv[i] = d > 0.f ? rsqrtf(d) : 0.f; }
}
__global__ __launch_bounds__(256)
void dse_gcn_scatter(float* __restrict__ out, const float* __restrict__ xw,
                     const int* __restrict__ src, const int* __restrict__ dst,
                     const float* __restrict__ w, const float* __restrict__ dinv, int E) {
  int e = blockIdx.x;
  if (e >= E) return;
  int s = src[e], d = dst[e];
  float coef = dinv[s] * w[e] * dinv[d];
  const float* xr = xw + (size_t)s * 512;
  float* orow = out + (size_t)d * 512;
  for (int c = threadIdx.x; c < 512; c += blockDim.x)
    atomicAdd(&orow[c], coef * xr[c]);
}
__global__ void dse_gcn_self(float* out, const float* __restrict__ xw,
                             const float* __restrict__ dinv, int n512) {
  int i = blockIdx.x * blockDim.x + threadIdx.x;
  if (i < n512) { int r = i >> 9; out[i] += dinv[r] * dinv[r] * xw[i]; }
}
__global__ void dse_gcn_finish(float* __restrict__ out, const float* __restrict__ base,
                               const float* __restrict__ gout, const float* __restrict__ bias,
                               const float* __restrict__ alpha, int n) {
  int i = blockIdx.x * blockDim.x + threadIdx.x;
  if (i < n) {
    int c = i & 511;
    out[i] = base[i] + alpha[0] * fmaxf(gout[i] + bias[c], 0.f);
  }
}
// se += llm_alpha[c] * (se_w2v + attn_alpha*se_attn)
__global__ void dse_se_emb(float* __restrict__ se, const float* __restrict__ w2v_t,
                           const float* __restrict__ attn_t, const float* __restrict__ la,
                           const float* __restrict__ aa, int n) {
  int i = blockIdx.x * blockDim.x + threadIdx.x;
  if (i < n) { int c = i & 511; se[i] += la[c] * (w2v_t[i] + aa[0] * attn_t[i]); }
}

// ===========================================================================
extern "C" void kernel_launch(void* const* d_in, const int* in_sizes, int n_in,
                              void* d_out, int out_size, void* d_ws, size_t ws_size,
                              hipStream_t stream) {
  (void)n_in; (void)out_size; (void)ws_size;

  // -------- input / param indices (leaf-wise insertion-order flattening) ----
  // 0 vecs, 1 fpts, 2 mpnns, 3 afps, 4 weaves, 5 nfs, 6 llm, 7 w2v,
  // 8 weight_drug, 9 weight_se, then params leaves, then edge indices.
  constexpr int MOL0 = 10, MOL1 = 16, FPT0 = 22, FPT1 = 28, GNN0 = 34;
  constexpr int CDS = 40, CSE = 61;             // 7 basic blocks x (w,g,b)
  constexpr int AT1 = 82, AT2 = 90, NRM1 = 94;
  constexpr int WD = 96, WA = 102, WN = 106, WNS = 116, WW = 122;
  constexpr int DGCN = 130, SGCN = 132, SEES = 134, LLMA = 135, ATTA = 136;
  constexpr int DSA = 137, SSA = 138, EID = 139, EIS = 140;

  auto F  = [&](int i) { return (const float*)d_in[i]; };
  auto IP = [&](int i) { return (const int*)d_in[i]; };
  const int E_D = in_sizes[8];   // 128000 drug edges
  const int E_S = in_sizes[9];   // 256000 se edges

  // -------- scratch layout --------------------------------------------------
  float* ws = (float*)d_ws;
  size_t cur = 0;
  auto alloc = [&](size_t n) { float* p = ws + cur; cur += n; return p; };

  float* vbuf = alloc(D_N * 64);   float* v2 = alloc(D_N * 64);
  float* g0 = alloc(D_N * 64);     float* g1 = alloc(D_N * 64);
  float* g2 = alloc(D_N * 64);     float* g3 = alloc(D_N * 64);
  float* fbf = alloc(D_N * 64);    float* f2 = alloc(D_N * 64);
  float* cses = alloc(D_N * 64);   float* mm = alloc(D_N * 64);
  float* gpre = alloc(D_N * 96);
  float* X1 = alloc((size_t)D_N * 4 * 64);
  float* pds = alloc((size_t)D_N * 4 * 64);
  float* tb  = alloc((size_t)D_N * 4 * 256);
  float* tb2 = alloc((size_t)D_N * 4 * 256);
  float* h1  = alloc((size_t)D_N * 2048);
  float* dsf = alloc((size_t)D_N * 512);
  float* daf = alloc((size_t)D_N * 512);
  float* h2  = alloc((size_t)D_N * 256);
  float* matk = alloc((size_t)3 * D_N * 64);
  float* kh = alloc((size_t)3 * D_N * 64);
  float* vh = alloc((size_t)3 * D_N * 64);
  float* qh = alloc(D_N * 64);  float* ctx = alloc(D_N * 64);  float* da = alloc(D_N * 64);
  float* xw   = alloc((size_t)S_N * 512);
  float* gout = alloc((size_t)S_N * 512);
  float* deg = alloc(S_N);  float* dinv = alloc(S_N);
  float* s128a = alloc((size_t)S_N * 128);
  float* s128b = alloc((size_t)S_N * 128);
  float* t512a = alloc((size_t)S_N * 512);
  float* t512b = alloc((size_t)S_N * 512);

  // d_out doubles as scratch until the final GEMM overwrites [0, 32M)
  float* ob = (float*)d_out;
  float* ic_ds  = ob;                       // 12,288,000 (phase A)
  float* dsb    = ob + 12288000;            // 16,384,000 (phase A)
  float* seconv = ob;                       //  4,096,000 (phase C)
  float* xout   = ob + 4096000;
  float* yout   = ob + 8192000;
  float* sew2v  = ob + 12288000;
  float* seattn = ob + 16384000;
  float* p768   = ob + 20480000;            // 6,144,000
  float* ic_se  = ob + 26624000;            // 3,072,000 (ends 29.7M < 32M)
  float* out_drug = ob + (size_t)D_N * S_N;           // 32,000,000
  float* out_se   = out_drug + (size_t)D_N * 512;     // 34,048,000

  auto blocks = [](long long n) { return dim3((unsigned)((n + 255) / 256)); };
  auto gemm = [&](const float* A, int lda, const float* B, int ldb, float* C, int ldc,
                  int M, int N, int K, const float* bias, int act) {
    dim3 g((N + GBN - 1) / GBN, (M + GBM - 1) / GBM);
    dse_wmma_gemm<<<g, 256, 0, stream>>>(A, lda, B, ldb, C, ldc, M, N, K, bias, act);
  };
  auto bn = [&](const float* x, int ldx, float* y, int ldy, int gi, int C, int M, int act) {
    dse_colbn<<<C, 256, 0, stream>>>(x, ldx, y, ldy, F(gi), F(gi + 1), M, act);
  };
  auto grudir = [&](const float* x, int I, int Wi, float* out, int coff) {
    gemm(x, I, F(Wi), I, gpre, 96, D_N, 96, I, F(Wi + 1), 0);
    dse_gru_act<<<blocks(D_N * 32), 256, 0, stream>>>(gpre, F(Wi + 2), out, D_N, 32, 64, coff);
  };

  // ============================ drug branch =================================
  // rnn_mol (2 bi-layers)
  grudir(F(0), 100, MOL0 + 0, vbuf, 0);  grudir(F(0), 100, MOL0 + 3, vbuf, 32);
  grudir(vbuf, 64, MOL1 + 0, v2, 0);     grudir(vbuf, 64, MOL1 + 3, v2, 32);
  // rnn_gnn on 4 graph embeddings
  { const float* gin[4] = {F(2), F(3), F(4), F(5)};
    float* gob[4] = {g0, g1, g2, g3};
    for (int t = 0; t < 4; ++t) {
      grudir(gin[t], 64, GNN0 + 0, gob[t], 0);
      grudir(gin[t], 64, GNN0 + 3, gob[t], 32);
    } }
  dse_mean4<<<blocks(D_N * 64), 256, 0, stream>>>(g0, g1, g2, g3, cses, D_N * 64);
  // rnn_fpt
  grudir(F(1), 1024, FPT0 + 0, fbf, 0);  grudir(F(1), 1024, FPT0 + 3, fbf, 32);
  grudir(fbf, 64, FPT1 + 0, f2, 0);      grudir(fbf, 64, FPT1 + 3, f2, 32);
  dse_mean3<<<blocks(D_N * 64), 256, 0, stream>>>(v2, cses, f2, mm, D_N * 64);

  // mcn -> inception(conv_ds), rows = D_N*4 (n,l), C grows 64 -> 256/branch
  dse_pack_mcn<<<blocks((size_t)D_N * 4 * 64), 256, 0, stream>>>(v2, cses, f2, mm, X1);
  const int R4 = D_N * 4;
  // b1
  gemm(X1, 64, F(CDS + 0), 64, tb, 256, R4, 256, 64, nullptr, 0);
  bn(tb, 256, tb, 256, CDS + 1, 256, R4, 1);
  dse_repack_ds<<<blocks((size_t)R4 * 256), 256, 0, stream>>>(tb, dsb, 0);
  // b3 = conv1 -> conv3
  gemm(X1, 64, F(CDS + 3), 64, tb, 256, R4, 256, 64, nullptr, 0);
  bn(tb, 256, tb, 256, CDS + 4, 256, R4, 1);
  dse_im2col3<<<blocks((size_t)R4 * 256), 256, 0, stream>>>(tb, ic_ds, R4, 4, 256);
  gemm(ic_ds, 768, F(CDS + 6), 768, tb2, 256, R4, 256, 768, nullptr, 0);
  bn(tb2, 256, tb2, 256, CDS + 7, 256, R4, 1);
  dse_repack_ds<<<blocks((size_t)R4 * 256), 256, 0, stream>>>(tb2, dsb, 256);
  // d = conv1 -> conv3 -> conv3
  gemm(X1, 64, F(CDS + 9), 64, tb, 256, R4, 256, 64, nullptr, 0);
  bn(tb, 256, tb, 256, CDS + 10, 256, R4, 1);
  dse_im2col3<<<blocks((size_t)R4 * 256), 256, 0, stream>>>(tb, ic_ds, R4, 4, 256);
  gemm(ic_ds, 768, F(CDS + 12), 768, tb2, 256, R4, 256, 768, nullptr, 0);
  bn(tb2, 256, tb2, 256, CDS + 13, 256, R4, 1);
  dse_im2col3<<<blocks((size_t)R4 * 256), 256, 0, stream>>>(tb2, ic_ds, R4, 4, 256);
  gemm(ic_ds, 768, F(CDS + 15), 768, tb, 256, R4, 256, 768, nullptr, 0);
  bn(tb, 256, tb, 256, CDS + 16, 256, R4, 1);
  dse_repack_ds<<<blocks((size_t)R4 * 256), 256, 0, stream>>>(tb, dsb, 512);
  // bp = avgpool3 -> conv1
  dse_avgpool3<<<blocks((size_t)R4 * 64), 256, 0, stream>>>(X1, pds, R4, 4, 64);
  gemm(pds, 64, F(CDS + 18), 64, tb, 256, R4, 256, 64, nullptr, 0);
  bn(tb, 256, tb, 256, CDS + 19, 256, R4, 1);
  dse_repack_ds<<<blocks((size_t)R4 * 256), 256, 0, stream>>>(tb, dsb, 768);

  // attn1 over mat = [v,cses,f], q = mm
  dse_pack3<<<blocks((size_t)3 * D_N * 64), 256, 0, stream>>>(v2, cses, f2, matk);
  gemm(mm, 64, F(AT1 + 0), 64, qh, 64, D_N, 64, 64, F(AT1 + 1), 0);
  gemm(matk, 64, F(AT1 + 2), 64, kh, 64, 3 * D_N, 64, 64, F(AT1 + 3), 0);
  gemm(matk, 64, F(AT1 + 4), 64, vh, 64, 3 * D_N, 64, 64, F(AT1 + 5), 0);
  dse_attn1<<<blocks(D_N * 2), 256, 0, stream>>>(qh, kh, vh, ctx);
  gemm(ctx, 64, F(AT1 + 6), 64, da, 64, D_N, 64, 64, F(AT1 + 7), 0);
  dse_rownorm64<<<D_N, 64, 0, stream>>>(da, da, F(NRM1), F(NRM1 + 1));

  // ds head: lin 4096->2048, bn+gelu, lin 2048->512
  gemm(dsb, 4096, F(WD + 0), 4096, h1, 2048, D_N, 2048, 4096, F(WD + 1), 0);
  bn(h1, 2048, h1, 2048, WD + 2, 2048, D_N, 1);
  gemm(h1, 2048, F(WD + 4), 2048, dsf, 512, D_N, 512, 2048, F(WD + 5), 0);
  // da head: gelu(lin 64->256), lin 256->512
  gemm(da, 64, F(WA + 0), 64, h2, 256, D_N, 256, 64, F(WA + 1), 1);
  gemm(h2, 256, F(WA + 2), 256, daf, 512, D_N, 512, 256, F(WA + 3), 0);
  dse_avg2<<<blocks((size_t)D_N * 512), 256, 0, stream>>>(dsf, daf, out_drug, D_N * 512);

  // drug GCN (self loops, symmetric norm) then drug += alpha*relu(gcn)
  gemm(out_drug, 512, F(DGCN), 512, xw, 512, D_N, 512, 512, nullptr, 0);
  dse_zero<<<blocks(D_N), 256, 0, stream>>>(deg, D_N);
  dse_zero<<<blocks((size_t)D_N * 512), 256, 0, stream>>>(gout, D_N * 512);
  dse_gcn_deg<<<blocks(E_D), 256, 0, stream>>>(deg, IP(EID) + E_D, F(8), E_D);
  dse_gcn_degself<<<blocks(D_N), 256, 0, stream>>>(deg, D_N);
  dse_gcn_dinv<<<blocks(D_N), 256, 0, stream>>>(dinv, deg, D_N);
  dse_gcn_scatter<<<dim3(E_D), 256, 0, stream>>>(gout, xw, IP(EID), IP(EID) + E_D, F(8), dinv, E_D);
  dse_gcn_self<<<blocks((size_t)D_N * 512), 256, 0, stream>>>(gout, xw, dinv, D_N * 512);
  dse_gcn_finish<<<blocks((size_t)D_N * 512), 256, 0, stream>>>(
      out_drug, out_drug, gout, F(DGCN + 1), F(DSA), D_N * 512);

  // ========================== side-effect branch ============================
  const float* llm = F(6);
  // inception(conv_se) on [C=768, L=8000]; rows are positions, llm is im2col-k1
  gemm(llm, 768, F(CSE + 0), 768, s128a, 128, S_N, 128, 768, nullptr, 0);
  bn(s128a, 128, seconv + 0, 512, CSE + 1, 128, S_N, 1);
  gemm(llm, 768, F(CSE + 3), 768, s128a, 128, S_N, 128, 768, nullptr, 0);
  bn(s128a, 128, s128a, 128, CSE + 4, 128, S_N, 1);
  dse_im2col3<<<blocks((size_t)S_N * 128), 256, 0, stream>>>(s128a, ic_se, S_N, S_N, 128);
  gemm(ic_se, 384, F(CSE + 6), 384, s128b, 128, S_N, 128, 384, nullptr, 0);
  bn(s128b, 128, seconv + 128, 512, CSE + 7, 128, S_N, 1);
  gemm(llm, 768, F(CSE + 9), 768, s128a, 128, S_N, 128, 768, nullptr, 0);
  bn(s128a, 128, s128a, 128, CSE + 10, 128, S_N, 1);
  dse_im2col3<<<blocks((size_t)S_N * 128), 256, 0, stream>>>(s128a, ic_se, S_N, S_N, 128);
  gemm(ic_se, 384, F(CSE + 12), 384, s128b, 128, S_N, 128, 384, nullptr, 0);
  bn(s128b, 128, s128b, 128, CSE + 13, 128, S_N, 1);
  dse_im2col3<<<blocks((size_t)S_N * 128), 256, 0, stream>>>(s128b, ic_se, S_N, S_N, 128);
  gemm(ic_se, 384, F(CSE + 15), 384, s128a, 128, S_N, 128, 384, nullptr, 0);
  bn(s128a, 128, seconv + 256, 512, CSE + 16, 128, S_N, 1);
  dse_avgpool3<<<blocks((size_t)S_N * 768), 256, 0, stream>>>(llm, p768, S_N, S_N, 768);
  gemm(p768, 768, F(CSE + 18), 768, s128a, 128, S_N, 128, 768, nullptr, 0);
  bn(s128a, 128, seconv + 384, 512, CSE + 19, 128, S_N, 1);

  // Wn chain on llm -> xout
  gemm(llm, 768, F(WN + 0), 768, t512a, 512, S_N, 512, 768, F(WN + 1), 1);
  gemm(t512a, 512, F(WN + 2), 512, t512b, 512, S_N, 512, 512, F(WN + 3), 1);
  gemm(t512b, 512, F(WN + 4), 512, t512a, 512, S_N, 512, 512, F(WN + 5), 0);
  bn(t512a, 512, t512a, 512, WN + 6, 512, S_N, 1);
  gemm(t512a, 512, F(WN + 8), 512, xout, 512, S_N, 512, 512, F(WN + 9), 0);
  // Wns chain on se_conv -> yout
  gemm(seconv, 512, F(WNS + 0), 512, t512a, 512, S_N, 512, 512, F(WNS + 1), 0);
  bn(t512a, 512, t512a, 512, WNS + 2, 512, S_N, 1);
  gemm(t512a, 512, F(WNS + 4), 512, yout, 512, S_N, 512, 512, F(WNS + 5), 0);
  dse_add2<<<blocks((size_t)S_N * 512), 256, 0, stream>>>(xout, yout, t512b, S_N * 512);  // se_llm
  // attn2 collapses to out_proj(v_proj(se_llm))
  gemm(t512b, 512, F(AT2 + 0), 512, t512a, 512, S_N, 512, 512, F(AT2 + 1), 0);
  gemm(t512a, 512, F(AT2 + 2), 512, seattn, 512, S_N, 512, 512, F(AT2 + 3), 0);
  // Wn_w2v chain -> sew2v
  gemm(F(7), 200, F(WW + 0), 200, t512a, 512, S_N, 512, 200, F(WW + 1), 1);
  gemm(t512a, 512, F(WW + 2), 512, yout, 512, S_N, 512, 512, F(WW + 3), 0);
  bn(yout, 512, yout, 512, WW + 4, 512, S_N, 1);
  gemm(yout, 512, F(WW + 6), 512, sew2v, 512, S_N, 512, 512, F(WW + 7), 0);

  // se GCN on `sees`, se = sees + alpha*relu(gcn)
  gemm(F(SEES), 512, F(SGCN), 512, xw, 512, S_N, 512, 512, nullptr, 0);
  dse_zero<<<blocks(S_N), 256, 0, stream>>>(deg, S_N);
  dse_zero<<<blocks((size_t)S_N * 512), 256, 0, stream>>>(gout, S_N * 512);
  dse_gcn_deg<<<blocks(E_S), 256, 0, stream>>>(deg, IP(EIS) + E_S, F(9), E_S);
  dse_gcn_degself<<<blocks(S_N), 256, 0, stream>>>(deg, S_N);
  dse_gcn_dinv<<<blocks(S_N), 256, 0, stream>>>(dinv, deg, S_N);
  dse_gcn_scatter<<<dim3(E_S), 256, 0, stream>>>(gout, xw, IP(EIS), IP(EIS) + E_S, F(9), dinv, E_S);
  dse_gcn_self<<<blocks((size_t)S_N * 512), 256, 0, stream>>>(gout, xw, dinv, S_N * 512);
  dse_gcn_finish<<<blocks((size_t)S_N * 512), 256, 0, stream>>>(
      out_se, F(SEES), gout, F(SGCN + 1), F(SSA), S_N * 512);
  // se_emb = se + llm_alpha * (se_w2v + attn_alpha*se_attn)
  dse_se_emb<<<blocks((size_t)S_N * 512), 256, 0, stream>>>(
      out_se, sew2v, seattn, F(LLMA), F(ATTA), S_N * 512);

  // ======================== outputs = drug @ se_emb^T ======================
  gemm(out_drug, 512, out_se, 512, (float*)d_out, S_N, D_N, S_N, 512, nullptr, 0);
}
